// PointFlow2DAdaptedCNF_52450140618809
// MI455X (gfx1250) — matrix operand order repaired and checked by reading, not compiled, not run on hardware
//
#include <hip/hip_runtime.h>
#include <hip/hip_bf16.h>

// ---------------------------------------------------------------------------
// PointFlow 2D CNF on MI455X (gfx1250, wave32, WMMA + TDM).
//
// Roofline: ~2.2 TFLOP total, ~3 MB live data -> compute bound. All 256-wide
// matmuls run on v_wmma_f32_16x16x32_f16 (f16 in, f32 acc). Layer-1 collapses
// to VALU because the 128-dim context contribution is per-batch constant
// (precomputed), and the JVP tangents enter layer-2 as silu'(pre1)*W1[0/1].
// Weight staging (W2/W3 ping-pong in LDS, 139 KB each) is offloaded to the
// Tensor Data Mover (tensor_load_to_lds, TENSORcnt), overlapping layer-1.
// ---------------------------------------------------------------------------

typedef _Float16 v16h __attribute__((ext_vector_type(16)));
typedef _Float16 v8h  __attribute__((ext_vector_type(8)));
typedef float    v8f  __attribute__((ext_vector_type(8)));
typedef unsigned int u32x4 __attribute__((ext_vector_type(4)));
typedef int          i32x4 __attribute__((ext_vector_type(4)));
typedef int          i32x8 __attribute__((ext_vector_type(8)));

#define HID   256
#define STR   272              // f16 row stride (544B = 34*16B -> aligned b128 loads)
#define WSTR  272
#define BUFH  (16 * STR)       // one 16-row activation buffer (f16 elements)
#define NWAVE 4                // 128-thread blocks, 64 points per block
#define DT    0.1f

struct F16x16 { v8h lo, hi; };

__device__ __forceinline__ v16h catf(v8h lo, v8h hi) {
  union { v16h v; v8h h[2]; } u; u.h[0] = lo; u.h[1] = hi; return u.v;
}

// A-fragment (16x32 f16): lane holds row m=lane&15; element e maps to
// k = (e>>3)*16 + half*8 + (e&7)  -> two contiguous v8h loads.
__device__ __forceinline__ F16x16 load_a(const _Float16* buf, int m, int half, int kt) {
  const _Float16* p = buf + m * STR + kt * 32 + half * 8;
  F16x16 f;
  f.lo = *(const v8h*)p;
  f.hi = *(const v8h*)(p + 16);
  return f;
}

// B-fragment (32x16 f16): lane holds column n=lane&15; k = (e>>3)*16 + half*8 + (e&7).
// Weights stored transposed [N][K] f16 with padded stride WSTR.
__device__ __forceinline__ v16h load_b(const _Float16* wlds, int nt, int lcol, int half, int kt) {
  const _Float16* p = wlds + (nt * 16 + lcol) * WSTR + kt * 32 + half * 8;
  return catf(*(const v8h*)p, *(const v8h*)(p + 16));
}

__device__ __forceinline__ v8f wmma_f16(v16h a, v16h b, v8f c) {
  return __builtin_amdgcn_wmma_f32_16x16x32_f16(false, a, false, b, (short)0, c, false, false);
}

__device__ __forceinline__ float sigm(float x) {
  return __builtin_amdgcn_rcpf(1.0f + __expf(-x));
}

// ---------------------------------------------------------------------------
// Tensor Data Mover: DMA one transposed weight matrix (256 rows x 272 f16 =
// 139,264 B, linear) from global into LDS. Descriptor per ISA 08 sec 8.3/8.4:
//   2D tile, data_size=3 (8 B units), dim0 = 544B/8 = 68, dim1 = 256 rows,
//   tensor_dim0_stride = 68 (contiguous), count=1, type=2 ("image").
// Issued wave-uniform by wave 0; completion via s_wait_tensorcnt + barrier.
// ---------------------------------------------------------------------------
__device__ __forceinline__ void tdm_load_weights(const _Float16* gsrc, unsigned lds_byte) {
  unsigned long long ga = (unsigned long long)(uintptr_t)gsrc;
  u32x4 g0 = { 1u,                                   // count=1, no gather
               lds_byte,                             // lds_addr [63:32]
               (unsigned)ga,                         // global_addr [95:64]
               (unsigned)((ga >> 32) & 0x01FFFFFFu) | 0x80000000u };  // addr hi | type=2
  i32x8 g1 = { 0x00030000,   // data_size=3 (8B), wg_mask=0
               0x00440000,   // tensor_dim0 = 68 (low 16 in [31:16])
               0x01000000,   // tensor_dim1 = 256 (low 16 in [31:16])
               0x00440000,   // tile_dim0 = 68 ([31:16])
               0x00000100,   // tile_dim1 = 256 ([15:0])
               68,           // tensor_dim0_stride = 68
               0, 0 };
  i32x4 gz = { 0, 0, 0, 0 };
#if __has_include(<hip/amd_detail/amd_gfx1250_TDM.h>)
  i32x8 gz8 = { 0, 0, 0, 0, 0, 0, 0, 0 };
  __builtin_amdgcn_tensor_load_to_lds(g0, g1, gz, gz, gz8, 0);
#else
  __builtin_amdgcn_tensor_load_to_lds(g0, g1, gz, gz, 0);
#endif
}

// ---------------------------------------------------------------------------
// Prep 1: W2/W3 (256x256 f32 row-major) -> f16 transposed [N][K], stride WSTR.
// ---------------------------------------------------------------------------
__global__ void cnf_prep_wt(const float* __restrict__ w2, const float* __restrict__ w3,
                            _Float16* __restrict__ wt2, _Float16* __restrict__ wt3) {
  int mat = blockIdx.x >> 8;
  int n   = blockIdx.x & 255;
  int k   = threadIdx.x;
  const float* W = mat ? w3 : w2;
  _Float16*   WT = mat ? wt3 : wt2;
  WT[n * WSTR + k] = (_Float16)W[k * HID + n];
}

// ---------------------------------------------------------------------------
// Prep 2: cbase[b][n] = b1[n] + sum_k ctx[b][k] * W1[3+k][n]   (per-batch).
// ---------------------------------------------------------------------------
__global__ void cnf_prep_ctx(const float* __restrict__ ctx, const float* __restrict__ w1,
                             const float* __restrict__ b1, float* __restrict__ cbase) {
  int b = blockIdx.x, n = threadIdx.x;
  float s = b1[n];
  for (int k = 0; k < 128; ++k)
    s = fmaf(ctx[b * 128 + k], w1[(3 + k) * HID + n], s);
  cbase[b * HID + n] = s;
}

// ---------------------------------------------------------------------------
// Main: 4096 blocks (64 batches x 64 groups of 64 points), 128 threads.
// Each wave owns 16 points. 10 Euler steps fused; W2/W3 DMA'd via TDM.
// ---------------------------------------------------------------------------
__global__ void __launch_bounds__(128, 1)
cnf_main(const float* __restrict__ x, const float* __restrict__ w1,
         const float* __restrict__ b2, const float* __restrict__ b3,
         const float* __restrict__ w4, const float* __restrict__ b4,
         const _Float16* __restrict__ wt2, const _Float16* __restrict__ wt3,
         const float* __restrict__ cbase, float* __restrict__ out_pts,
         float* __restrict__ partial) {
  __shared__ __align__(16) _Float16 WgtLds[HID * WSTR];           // 139264 B
  __shared__ __align__(16) _Float16 Stage[NWAVE * 5 * BUFH];      // 174080 B
  __shared__ float W4s[HID * 2];                                  //   2048 B
  __shared__ float w1r[3][HID];                                   //   3072 B
  __shared__ __align__(16) _Float16 w1rh[2][HID];                 //   1024 B
  __shared__ float cb[HID];                                       //   1024 B
  __shared__ float b2s[HID], b3s[HID];                            //   2048 B
  __shared__ float ptsb[64][2];                                   //    512 B
  __shared__ float ldb[64];                                       //    256 B

  const int tid  = threadIdx.x;
  const int wv   = tid >> 5;
  const int lane = tid & 31;
  const int lcol = lane & 15;
  const int half = lane >> 4;
  const int bB   = blockIdx.x >> 6;        // batch
  const int p0   = (blockIdx.x & 63) * 64; // point group base

  const unsigned wgt_lds_addr = (unsigned)(uintptr_t)&WgtLds[0];

  // ---- one-time block init ----
  {
    int p = tid >> 1, c = tid & 1;
    ptsb[p][c] = x[((size_t)(bB * 4096 + p0 + p)) * 2 + c];
    if (tid < 64) ldb[tid] = 0.0f;
    for (int i = tid; i < 512; i += 128) W4s[i] = w4[i];
    for (int i = tid; i < 768; i += 128) ((float*)w1r)[i] = w1[i];
    for (int i = tid; i < 512; i += 128) ((_Float16*)w1rh)[i] = (_Float16)w1[i];
    for (int i = tid; i < 256; i += 128) {
      cb[i]  = cbase[bB * HID + i];
      b2s[i] = b2[i];
      b3s[i] = b3[i];
    }
  }
  const float b4_0 = b4[0], b4_1 = b4[1];
  __syncthreads();

  _Float16* INH = Stage + (wv * 5 + 0) * BUFH;
  _Float16* INS = Stage + (wv * 5 + 1) * BUFH;
  _Float16* OH  = Stage + (wv * 5 + 2) * BUFH;
  _Float16* OU  = Stage + (wv * 5 + 3) * BUFH;
  _Float16* OW  = Stage + (wv * 5 + 4) * BUFH;

#pragma unroll 1
  for (int step = 0; step < 10; ++step) {
    const float t = (float)step * DT;

    // ---- kick TDM for W2; DMA overlaps with the layer-1 VALU phase ----
    if (wv == 0) tdm_load_weights(wt2, wgt_lds_addr);

    // ---- layer 1 (VALU): pre1 = x0*W1r0 + x1*W1r1 + t*W1r2 + cbase ----
    {
      int m  = lane >> 1;
      int nh = lane & 1;
      float x0 = ptsb[wv * 16 + m][0];
      float x1 = ptsb[wv * 16 + m][1];
      for (int n0 = 0; n0 < 128; ++n0) {
        int n = nh * 128 + n0;
        float pre = fmaf(x0, w1r[0][n], fmaf(x1, w1r[1][n], fmaf(t, w1r[2][n], cb[n])));
        float sg = sigm(pre);
        INH[m * STR + n] = (_Float16)(pre * sg);
        INS[m * STR + n] = (_Float16)(sg * (1.0f + pre * (1.0f - sg)));
      }
    }
    if (wv == 0) __builtin_amdgcn_s_wait_tensorcnt(0);
    __syncthreads();

    // ---- layer 2: three streams (h1, s1*W1r0, s1*W1r1) @ W2, fused silu'/scale ----
    for (int nc = 0; nc < 4; ++nc) {
      v8f accF[4], accU[4], accW[4];
      const v8f z = {};
#pragma unroll
      for (int nn = 0; nn < 4; ++nn) { accF[nn] = z; accU[nn] = z; accW[nn] = z; }
#pragma unroll
      for (int kt = 0; kt < 8; ++kt) {
        F16x16 fH = load_a(INH, lcol, half, kt);
        F16x16 fS = load_a(INS, lcol, half, kt);
        v8h w0lo = *(const v8h*)&w1rh[0][kt * 32 + half * 8];
        v8h w0hi = *(const v8h*)&w1rh[0][kt * 32 + half * 8 + 16];
        v8h w1lo = *(const v8h*)&w1rh[1][kt * 32 + half * 8];
        v8h w1hi = *(const v8h*)&w1rh[1][kt * 32 + half * 8 + 16];
        v16h aH = catf(fH.lo, fH.hi);
        v16h aU = catf(fS.lo * w0lo, fS.hi * w0hi);
        v16h aW = catf(fS.lo * w1lo, fS.hi * w1hi);
#pragma unroll
        for (int nn = 0; nn < 4; ++nn) {
          v16h bF = load_b(WgtLds, nc * 4 + nn, lcol, half, kt);
          accF[nn] = wmma_f16(aH, bF, accF[nn]);
          accU[nn] = wmma_f16(aU, bF, accU[nn]);
          accW[nn] = wmma_f16(aW, bF, accW[nn]);
        }
      }
#pragma unroll
      for (int nn = 0; nn < 4; ++nn) {
        int col = (nc * 4 + nn) * 16 + lcol;
        float bb = b2s[col];
#pragma unroll
        for (int r = 0; r < 8; ++r) {
          float pre = accF[nn][r] + bb;
          float sg  = sigm(pre);
          float s2  = sg * (1.0f + pre * (1.0f - sg));
          int row   = half * 8 + r;
          OH[row * STR + col] = (_Float16)(pre * sg);
          OU[row * STR + col] = (_Float16)(s2 * accU[nn][r]);
          OW[row * STR + col] = (_Float16)(s2 * accW[nn][r]);
        }
      }
    }
    __syncthreads();

    // ---- TDM for W3 (all waves are past their W2 reads) ----
    if (wv == 0) {
      tdm_load_weights(wt3, wgt_lds_addr);
      __builtin_amdgcn_s_wait_tensorcnt(0);
    }
    __syncthreads();

    // ---- layer 3 @ W3, fold W4 (256->2) into epilogue, no output staging ----
    float v0[8], v1[8], j0a[8], j1a[8];
#pragma unroll
    for (int r = 0; r < 8; ++r) { v0[r] = 0.f; v1[r] = 0.f; j0a[r] = 0.f; j1a[r] = 0.f; }

    for (int nc = 0; nc < 4; ++nc) {
      v8f accF[4], accU[4], accW[4];
      const v8f z = {};
#pragma unroll
      for (int nn = 0; nn < 4; ++nn) { accF[nn] = z; accU[nn] = z; accW[nn] = z; }
#pragma unroll
      for (int kt = 0; kt < 8; ++kt) {
        F16x16 fH = load_a(OH, lcol, half, kt);
        F16x16 fU = load_a(OU, lcol, half, kt);
        F16x16 fW = load_a(OW, lcol, half, kt);
        v16h aH = catf(fH.lo, fH.hi);
        v16h aU = catf(fU.lo, fU.hi);
        v16h aW = catf(fW.lo, fW.hi);
#pragma unroll
        for (int nn = 0; nn < 4; ++nn) {
          v16h bF = load_b(WgtLds, nc * 4 + nn, lcol, half, kt);
          accF[nn] = wmma_f16(aH, bF, accF[nn]);
          accU[nn] = wmma_f16(aU, bF, accU[nn]);
          accW[nn] = wmma_f16(aW, bF, accW[nn]);
        }
      }
#pragma unroll
      for (int nn = 0; nn < 4; ++nn) {
        int col = (nc * 4 + nn) * 16 + lcol;
        float bb  = b3s[col];
        float w40 = W4s[2 * col];
        float w41 = W4s[2 * col + 1];
#pragma unroll
        for (int r = 0; r < 8; ++r) {
          float pre = accF[nn][r] + bb;
          float sg  = sigm(pre);
          float h3  = pre * sg;
          float s3  = sg * (1.0f + pre * (1.0f - sg));
          float ju  = s3 * accU[nn][r];
          float jw  = s3 * accW[nn][r];
          v0[r]  = fmaf(h3, w40, v0[r]);
          v1[r]  = fmaf(h3, w41, v1[r]);
          j0a[r] = fmaf(ju, w40, j0a[r]);
          j1a[r] = fmaf(jw, w41, j1a[r]);
        }
      }
    }

    // ---- reduce over 16 columns (xor-shuffle within each 16-lane half) ----
#pragma unroll
    for (int off = 1; off < 16; off <<= 1) {
#pragma unroll
      for (int r = 0; r < 8; ++r) {
        v0[r]  += __shfl_xor(v0[r],  off, 32);
        v1[r]  += __shfl_xor(v1[r],  off, 32);
        j0a[r] += __shfl_xor(j0a[r], off, 32);
        j1a[r] += __shfl_xor(j1a[r], off, 32);
      }
    }

    // ---- Euler update: lanes (r) and (16+r) own rows r and 8+r ----
#pragma unroll
    for (int r = 0; r < 8; ++r) {
      if (lcol == r) {
        int mm = wv * 16 + half * 8 + r;
        ptsb[mm][0] += DT * (v0[r] + b4_0);
        ptsb[mm][1] += DT * (v1[r] + b4_1);
        ldb[mm]     += DT * (j0a[r] + j1a[r]);
      }
    }
    __syncthreads();   // protects ptsb/ldb and next-step weight swap
  }

  // ---- write final points + per-block logdet partial ----
  {
    int p = tid >> 1, c = tid & 1;
    out_pts[((size_t)(bB * 4096 + p0 + p)) * 2 + c] = ptsb[p][c];
  }
  if (tid == 0) {
    float s = 0.0f;
    for (int j = 0; j < 64; ++j) s += ldb[j];
    partial[blockIdx.x] = s;
  }
}

// ---------------------------------------------------------------------------
// Final deterministic logdet reduction: 64 partials per batch -> d_out tail.
// ---------------------------------------------------------------------------
__global__ void cnf_reduce(const float* __restrict__ partial, float* __restrict__ out_ld) {
  if (threadIdx.x == 0) {
    float s = 0.0f;
    for (int j = 0; j < 64; ++j) s += partial[blockIdx.x * 64 + j];
    out_ld[blockIdx.x] = s;
  }
}

extern "C" void kernel_launch(void* const* d_in, const int* in_sizes, int n_in,
                              void* d_out, int out_size, void* d_ws, size_t ws_size,
                              hipStream_t stream) {
  const float* x   = (const float*)d_in[0];
  const float* ctx = (const float*)d_in[1];
  const float* w1  = (const float*)d_in[2];
  const float* b1  = (const float*)d_in[3];
  const float* w2  = (const float*)d_in[4];
  const float* b2  = (const float*)d_in[5];
  const float* w3  = (const float*)d_in[6];
  const float* b3  = (const float*)d_in[7];
  const float* w4  = (const float*)d_in[8];
  const float* b4  = (const float*)d_in[9];

  // workspace: WT2(f16) | WT3(f16) | cbase(f32) | partial(f32)  ~= 352 KB
  _Float16* wt2   = (_Float16*)d_ws;
  _Float16* wt3   = wt2 + HID * WSTR;
  float*    cb    = (float*)(wt3 + HID * WSTR);
  float*    part  = cb + 64 * HID;

  cnf_prep_wt <<<512, 256, 0, stream>>>(w2, w3, wt2, wt3);
  cnf_prep_ctx<<<64, 256, 0, stream>>>(ctx, w1, b1, cb);
  cnf_main    <<<4096, 128, 0, stream>>>(x, w1, b2, b3, w4, b4, wt2, wt3, cb,
                                         (float*)d_out, part);
  cnf_reduce  <<<64, 32, 0, stream>>>(part, (float*)d_out + 64 * 4096 * 2);
}
